// TernaryLinear_46084999086226
// MI455X (gfx1250) — matrix operand order repaired and checked by reading, not compiled, not run on hardware
//
#include <hip/hip_runtime.h>
#include <stdint.h>
#include <stddef.h>

// ---------------------------------------------------------------------------
// TernaryLinear for MI455X (gfx1250, wave32, WMMA).
// out[b,s,o] = sum_k x[b,s,k] * tern(W[o,k]) * scale + bias[o]
// Strategy: ternarize W -> f16 {-1,0,+1} (exact), convert x -> f16,
// f16 WMMA GEMM with f32 accumulate, scale applied in f32 epilogue.
// Compute-bound: ~0.55 TFLOP vs ~0.6 GB HBM traffic; f16 WMMA is the
// fastest path that preserves accuracy (weights are exact in f16, scale
// applied in f32). x_f16 (134 MB) + W_f16 (33 MB) fit in the 192 MB L2.
// ---------------------------------------------------------------------------

typedef _Float16 f16;
typedef __attribute__((ext_vector_type(16))) _Float16 v16h;
typedef __attribute__((ext_vector_type(8)))  _Float16 v8h;
typedef __attribute__((ext_vector_type(8)))  float    v8f;
typedef __attribute__((ext_vector_type(4)))  int      v4i;

#define D_IN   4096
#define D_OUT  4096
#define M_ROWS 16384
#define BK     32
#define NKSTEP (D_IN / BK)   // 128 K-steps
#define THRESH 0.5f

// Async global->LDS path only if the gfx1250 builtin exists in this toolchain.
#if defined(__HIP_DEVICE_COMPILE__) && __has_builtin(__builtin_amdgcn_global_load_async_to_lds_b128)
#define USE_ASYNC 1
#else
#define USE_ASYNC 0
#endif

__device__ __forceinline__ void wait_async_le2() {
#if defined(__HIP_DEVICE_COMPILE__) && __has_builtin(__builtin_amdgcn_s_wait_asynccnt)
  __builtin_amdgcn_s_wait_asynccnt(2);
#elif defined(__HIP_DEVICE_COMPILE__)
  asm volatile("s_wait_asynccnt 2" ::: "memory");
#endif
}
__device__ __forceinline__ void wait_async_le0() {
#if defined(__HIP_DEVICE_COMPILE__) && __has_builtin(__builtin_amdgcn_s_wait_asynccnt)
  __builtin_amdgcn_s_wait_asynccnt(0);
#elif defined(__HIP_DEVICE_COMPILE__)
  asm volatile("s_wait_asynccnt 0" ::: "memory");
#endif
}

// ---------------- scale computation -----------------------------------------

__global__ void zero_stats_k(float* stats) {
  stats[0] = 0.0f;                  // ssum
  ((unsigned int*)stats)[1] = 0u;   // cnt
}

__global__ void reduce_absgt_k(const float* __restrict__ w, float* stats, int n) {
  __shared__ float    ssum[256];
  __shared__ unsigned scnt[256];
  float s = 0.0f;
  unsigned c = 0;
  for (int i = blockIdx.x * blockDim.x + threadIdx.x; i < n;
       i += gridDim.x * blockDim.x) {
    float a = fabsf(w[i]);
    if (a > THRESH) { s += a; ++c; }
  }
  ssum[threadIdx.x] = s;
  scnt[threadIdx.x] = c;
  __syncthreads();
  for (int off = 128; off > 0; off >>= 1) {
    if ((int)threadIdx.x < off) {
      ssum[threadIdx.x] += ssum[threadIdx.x + off];
      scnt[threadIdx.x] += scnt[threadIdx.x + off];
    }
    __syncthreads();
  }
  if (threadIdx.x == 0) {
    atomicAdd(&stats[0], ssum[0]);
    atomicAdd(((unsigned int*)stats) + 1, scnt[0]);
  }
}

__global__ void finalize_scale_k(float* stats) {
  unsigned c = ((unsigned int*)stats)[1];
  stats[2] = (c > 0u) ? (stats[0] / (float)c) : 1.0f;
}

// ---------------- quantization ----------------------------------------------

__global__ void ternarize_w_k(const float* __restrict__ w,
                              uint16_t* __restrict__ wq, int n) {
  int i = blockIdx.x * blockDim.x + threadIdx.x;
  if (i < n) {
    float v = w[i];
    f16 t = (f16)((v > THRESH) ? 1.0f : ((v < -THRESH) ? -1.0f : 0.0f));
    union { f16 h; uint16_t u; } cv;
    cv.h = t;
    wq[i] = cv.u;
  }
}

__global__ void convert_x4_k(const float* __restrict__ x,
                             uint16_t* __restrict__ xq, int n4) {
  int i = blockIdx.x * blockDim.x + threadIdx.x;
  if (i < n4) {
    float4 v = ((const float4*)x)[i];
    union { f16 h[4]; uint2 u; } cv;
    cv.h[0] = (f16)v.x; cv.h[1] = (f16)v.y;
    cv.h[2] = (f16)v.z; cv.h[3] = (f16)v.w;
    ((uint2*)xq)[i] = cv.u;
  }
}

// ---------------- WMMA GEMM -------------------------------------------------
// Block: 256 threads = 8 wave32 waves. Output tile 128(M) x 128(N).
// Wave w owns M rows [mBase+16w, +16), all 8 N sub-tiles -> 8 x v8f acc.
// B tile (128 rows of W x BK) staged in LDS, double-buffered (async path).

__global__ __launch_bounds__(256) void ternary_gemm_wmma(
    const uint16_t* __restrict__ xq_u, const uint16_t* __restrict__ wq_u,
    const float* __restrict__ bias, const float* __restrict__ stats,
    float* __restrict__ out) {
  const f16* xq = (const f16*)xq_u;
  f16* wq = (f16*)wq_u;  // non-const simplifies addrspace casts for async copy

  __shared__ __align__(32) f16 Bs[2][128 * BK];  // 2 x 8 KB

  const int t     = threadIdx.x;
  const int lane  = t & 31;
  const int wave  = t >> 5;            // 0..7
  const int nBase = blockIdx.x * 128;
  const int mBase = blockIdx.y * 128;
  const int laneM = lane & 15;
  const int laneH = lane >> 4;         // 0 or 1

  // A fragment source: contiguous k order, matching B — lanes 0-15: K 0..15,
  // lanes 16-31: K 16..31. Two b128 loads at +0 and +8 halves.
  const f16* aPtr =
      xq + (size_t)(mBase + wave * 16 + laneM) * D_IN + laneH * 16;

  // B staging: 512 x 16B chunks per tile, 2 per thread.
  const int c0 = t, c1 = t + 256;
  const int r0 = c0 >> 2, cc0 = c0 & 3;
  const int r1 = c1 >> 2, cc1 = c1 & 3;
  f16* src0 = wq + (size_t)(nBase + r0) * D_IN + cc0 * 8;
  f16* src1 = wq + (size_t)(nBase + r1) * D_IN + cc1 * 8;
  f16* dst0 = &Bs[0][r0 * BK + cc0 * 8];
  f16* dst1 = &Bs[0][r1 * BK + cc1 * 8];

  v8f acc[8];
#pragma unroll
  for (int i = 0; i < 8; ++i) acc[i] = (v8f){0, 0, 0, 0, 0, 0, 0, 0};

  auto compute_tile = [&](int buf, int k0) {
    v8h a0 = *(const v8h*)(aPtr + k0);
    v8h a1 = *(const v8h*)(aPtr + k0 + 8);
    union { v16h v; v8h h[2]; } au;
    au.h[0] = a0;
    au.h[1] = a1;
    // B operand, same contiguous k order as A: lane&15 = N column (row of W), lanes 0-15 hold
    // K 0..15, lanes 16-31 hold K 16..31 -> one contiguous 32B read.
    const f16* bp = &Bs[buf][laneM * BK + laneH * 16];
#pragma unroll
    for (int ns = 0; ns < 8; ++ns) {
      v16h bv = *(const v16h*)(bp + ns * 16 * BK);
      acc[ns] = __builtin_amdgcn_wmma_f32_16x16x32_f16(
          false, au.v, false, bv, (short)0, acc[ns], false, false);
    }
  };

#if USE_ASYNC
  auto issue = [&](int buf, int k0) {
    f16* d0 = dst0 + buf * (128 * BK);
    f16* d1 = dst1 + buf * (128 * BK);
    // Builtin signature (from compiler diagnostic): arg0 is int4* in the
    // global (__device__) address space; arg1 is the LDS destination.
    __builtin_amdgcn_global_load_async_to_lds_b128(
        (__attribute__((address_space(1))) v4i*)(void*)(src0 + k0),
        (__attribute__((address_space(3))) v4i*)(void*)d0, 0, 0);
    __builtin_amdgcn_global_load_async_to_lds_b128(
        (__attribute__((address_space(1))) v4i*)(void*)(src1 + k0),
        (__attribute__((address_space(3))) v4i*)(void*)d1, 0, 0);
  };

  issue(0, 0);  // prologue stage
  for (int kt = 0; kt < NKSTEP; ++kt) {
    const int buf = kt & 1;
    if (kt + 1 < NKSTEP) {
      issue(buf ^ 1, (kt + 1) * BK);  // overlap next tile's loads
      wait_async_le2();               // our 2 loads for tile kt are done
    } else {
      wait_async_le0();
    }
    __syncthreads();                  // whole tile visible in LDS
    compute_tile(buf, kt * BK);
    if (kt + 1 < NKSTEP) __syncthreads();  // done reading before overwrite
  }
#else
  for (int kt = 0; kt < NKSTEP; ++kt) {
    const int k0 = kt * BK;
    __syncthreads();  // everyone finished reading the previous tile
    *(uint4*)dst0 = *(const uint4*)(src0 + k0);
    *(uint4*)dst1 = *(const uint4*)(src1 + k0);
    __syncthreads();
    compute_tile(0, k0);
  }
#endif

  // Epilogue: C/D 16x16 f32 layout — lane N = lane&15, VGPR j -> M = 8*laneH + j.
  const float scale = stats[2];
  const int nCol = nBase + laneM;
  const int mTop = mBase + wave * 16 + laneH * 8;
#pragma unroll
  for (int ns = 0; ns < 8; ++ns) {
    const int n = nCol + ns * 16;
    const float b = bias[n];
#pragma unroll
    for (int j = 0; j < 8; ++j) {
      out[(size_t)(mTop + j) * D_OUT + n] = acc[ns][j] * scale + b;
    }
  }
}

// ---------------- host launcher ---------------------------------------------

extern "C" void kernel_launch(void* const* d_in, const int* in_sizes, int n_in,
                              void* d_out, int out_size, void* d_ws,
                              size_t ws_size, hipStream_t stream) {
  (void)in_sizes; (void)n_in; (void)out_size; (void)ws_size;

  const float* x    = (const float*)d_in[0];
  const float* w    = (const float*)d_in[1];
  const float* bias = (const float*)d_in[2];
  float* out = (float*)d_out;

  // Workspace layout: [0..255] stats {ssum, cnt(u32), scale},
  // then W_f16 (32 MB), then x_f16 (128 MB). ~168 MB total.
  float*    stats = (float*)d_ws;
  uint16_t* wq = (uint16_t*)((char*)d_ws + 256);
  uint16_t* xq = (uint16_t*)((char*)d_ws + 256 +
                             (size_t)D_OUT * D_IN * sizeof(uint16_t));

  zero_stats_k<<<1, 1, 0, stream>>>(stats);
  reduce_absgt_k<<<2048, 256, 0, stream>>>(w, stats, D_OUT * D_IN);
  finalize_scale_k<<<1, 1, 0, stream>>>(stats);
  ternarize_w_k<<<(D_OUT * D_IN) / 256, 256, 0, stream>>>(w, wq, D_OUT * D_IN);
  convert_x4_k<<<(M_ROWS * D_IN / 4) / 256, 256, 0, stream>>>(
      x, xq, M_ROWS * D_IN / 4);
  ternary_gemm_wmma<<<dim3(D_OUT / 128, M_ROWS / 128), 256, 0, stream>>>(
      xq, wq, bias, stats, out);
}
